// Decoder_44702019617603
// MI455X (gfx1250) — matrix operand order repaired and checked by reading, not compiled, not run on hardware
//
#include <hip/hip_runtime.h>
#include <hip/hip_bf16.h>
#include <stdint.h>

// ---------------------------------------------------------------------------
// Types / helpers
// ---------------------------------------------------------------------------
typedef __attribute__((ext_vector_type(16))) __bf16 v16bf;
typedef __attribute__((ext_vector_type(8)))  float  v8f;
typedef __attribute__((ext_vector_type(4)))  int    v4i_;

#define B_   2048
#define S_   128
#define V_   128
#define E_   300
#define H_   512
#define KIH  832   // 812 padded to multiple of 64
#define EPAD 320   // 300 padded to multiple of 64

// gfx1250 async global->LDS path (ASYNCcnt-tracked), guarded so the file
// compiles even if the toolchain lacks the builtin.
#if __has_builtin(__builtin_amdgcn_global_load_async_to_lds_b128)
#define USE_ASYNC_LDS 1
#endif

__device__ __forceinline__ void wait_async0() {
#ifdef USE_ASYNC_LDS
#if __has_builtin(__builtin_amdgcn_s_wait_asynccnt)
  __builtin_amdgcn_s_wait_asynccnt(0);
#else
  asm volatile("s_wait_asynccnt 0x0" ::: "memory");
#endif
#endif
}

// 16B global -> LDS copy: async (ASYNCcnt) when available, else sync via regs
__device__ __forceinline__ void cp_b128(const void* g, void* l) {
#ifdef USE_ASYNC_LDS
  __builtin_amdgcn_global_load_async_to_lds_b128(
      (__attribute__((address_space(1))) v4i_*)g,
      (__attribute__((address_space(3))) v4i_*)l, 0, 0);
#else
  *(uint4*)l = *(const uint4*)g;
#endif
}

__device__ __forceinline__ __bf16 f2bf(float f) {
  union { float f; uint32_t u; } a; a.f = f;
  uint32_t r = a.u + 0x7FFFu + ((a.u >> 16) & 1u);   // round-to-nearest-even
  union { uint16_t s; __bf16 b; } o; o.s = (uint16_t)(r >> 16); return o.b;
}
__device__ __forceinline__ float bf2f(__bf16 b) {
  union { __bf16 b; uint16_t s; } i; i.b = b;
  union { uint32_t u; float f; } o; o.u = ((uint32_t)i.s) << 16; return o.f;
}

union BFrag { v16bf v; uint4 q[2]; };

// ---------------------------------------------------------------------------
// f32 -> bf16 convert with zero padding (used for weights and activations)
// ---------------------------------------------------------------------------
__global__ void convert_pad_kernel(const float* __restrict__ src,
                                   __bf16* __restrict__ dst,
                                   int rows, int cols, int ldd, int rowsPad) {
  int idx = blockIdx.x * blockDim.x + threadIdx.x;
  int total = rowsPad * ldd;
  if (idx >= total) return;
  int r = idx / ldd, c = idx % ldd;
  float v = (r < rows && c < cols) ? src[(size_t)r * cols + c] : 0.0f;
  dst[idx] = f2bf(v);
}

// ---------------------------------------------------------------------------
// Generic bf16 WMMA GEMM:  out[M,N] = A[M,K] @ W[N,K]^T + bias
// Block = 256 threads (8 waves). Tile: M=128, N=64, K-chunk=64.
// Double-buffered LDS staging; async global->LDS when available.
// Wave w owns m-block w (16 rows) and 4 n-blocks (4 accumulators).
// K and lda/ldw must be multiples of 64/8; M multiple of 128.
// ---------------------------------------------------------------------------
__global__ __launch_bounds__(256) void gemm_bf16_kernel(
    const __bf16* __restrict__ A, int lda,
    const __bf16* __restrict__ W, int ldw,
    const float* __restrict__ bias,
    float* __restrict__ out, int ldo,
    int N, int K) {
  __shared__ __bf16 As[2][128][72];   // +8 pad to break bank conflicts
  __shared__ __bf16 Ws[2][64][72];

  const int tid  = threadIdx.x;
  const int lane = tid & 31;
  const int wave = tid >> 5;
  const int half = lane >> 4;    // 0/1
  const int l16  = lane & 15;
  const int m0   = blockIdx.x * 128;
  const int n0   = blockIdx.y * 64;

  v8f zero = {0,0,0,0,0,0,0,0};
  v8f acc[4] = {zero, zero, zero, zero};

  auto stage = [&](int buf, int k0) {
    // A tile 128x64 (8 x b128 per row), W tile 64x64
    for (int i = tid; i < 128 * 8; i += 256) {
      int r = i >> 3, c8 = (i & 7) * 8;
      cp_b128(A + (size_t)(m0 + r) * lda + k0 + c8, &As[buf][r][c8]);
    }
    for (int i = tid; i < 64 * 8; i += 256) {
      int r = i >> 3, c8 = (i & 7) * 8;
      cp_b128(W + (size_t)(n0 + r) * ldw + k0 + c8, &Ws[buf][r][c8]);
    }
  };

  const int nk = K >> 6;
  stage(0, 0);
  for (int t = 0; t < nk; t++) {
    const int cur = t & 1;
    wait_async0();        // tile t landed (own ASYNCcnt)
    __syncthreads();      // ... across all waves; also fences buf reuse
    if (t + 1 < nk) stage(cur ^ 1, (t + 1) * 64);

    #pragma unroll
    for (int kk = 0; kk < 2; kk++) {
      // Preload A fragment + all 4 B fragments, then 4 back-to-back WMMAs
      BFrag af;
      const __bf16* pa = &As[cur][wave * 16 + l16][kk * 32 + half * 8];
      af.q[0] = *(const uint4*)pa;
      af.q[1] = *(const uint4*)(pa + 16);
      BFrag bfr[4];
      #pragma unroll
      for (int j = 0; j < 4; j++) {
        const __bf16* pb = &Ws[cur][j * 16 + l16][kk * 32 + half * 8];
        bfr[j].q[0] = *(const uint4*)pb;
        bfr[j].q[1] = *(const uint4*)(pb + 16);
      }
      #pragma unroll
      for (int j = 0; j < 4; j++)
        acc[j] = __builtin_amdgcn_wmma_f32_16x16x32_bf16(
            false, af.v, false, bfr[j].v, (short)0, acc[j], false, false);
    }
  }

  // Epilogue: C/D layout -> lane l, vgpr v: m = half*8+v, n = l16
  #pragma unroll
  for (int j = 0; j < 4; j++) {
    int n = n0 + j * 16 + l16;
    if (n >= N) continue;
    float bv = bias ? bias[n] : 0.0f;
    #pragma unroll
    for (int v = 0; v < 8; v++) {
      int m = m0 + wave * 16 + half * 8 + v;
      out[(size_t)m * ldo + n] = acc[j][v] + bv;
    }
  }
}

// ---------------------------------------------------------------------------
// Fused attention: one workgroup per batch row b.
//   score = tanh(enc[b] @ W1^T + W1_b + t2[b])   (never materialized)
//   s2    = score @ V_w^T + V_b ; attw = softmax_S(s2)
//   ctx[b] = sum_s attw[s] * enc[b,s,:]
// enc[b] (128x512) is converted f32->bf16 into LDS once and reused for both
// the WMMA GEMM (A operand) and the context reduction.
// ---------------------------------------------------------------------------
#define ENC_LDA 520   // 512 + 8 pad (bf16 elems)

__global__ __launch_bounds__(256) void attn_fused_kernel(
    const float* __restrict__ enc,     // (B,128,512) f32
    const __bf16* __restrict__ W1,     // (512,512) bf16
    const float* __restrict__ W1b,     // (512)
    const float* __restrict__ t2,      // (B,512) f32
    const float* __restrict__ Vw,      // (512)
    const float* __restrict__ Vb,      // (1)
    float* __restrict__ ctx) {         // (B,512) f32
  extern __shared__ char smem[];
  __bf16* encs = (__bf16*)smem;                       // 128 x ENC_LDA
  float*  s2   = (float*)(smem + 128 * ENC_LDA * 2);  // 128
  float*  red  = s2 + 128;                            // 128
  float*  attw = red + 128;                           // 128

  const int b    = blockIdx.x;
  const int tid  = threadIdx.x;
  const int lane = tid & 31;
  const int wave = tid >> 5;
  const int half = lane >> 4;
  const int l16  = lane & 15;
  const float* encb = enc + (size_t)b * (S_ * H_);

  // Load + convert enc[b] to LDS bf16 (single HBM pass over enc)
  for (int i = tid; i < S_ * (H_ / 4); i += 256) {
    int r = i >> 7;            // 128 float4 per row
    int c = (i & 127) * 4;
    float4 v = *(const float4*)(encb + r * H_ + c);
    union { __bf16 h[4]; uint2 u; } pk;
    pk.h[0] = f2bf(v.x); pk.h[1] = f2bf(v.y);
    pk.h[2] = f2bf(v.z); pk.h[3] = f2bf(v.w);
    *(uint2*)(&encs[r * ENC_LDA + c]) = pk.u;
  }
  if (tid < 128) s2[tid] = Vb[0];
  __syncthreads();

  // GEMM + tanh + V_w contraction. Wave w owns m-block w; loop 32 n-blocks.
  const float* t2b = t2 + (size_t)b * H_;
  for (int nb = 0; nb < 32; nb++) {
    int n = nb * 16 + l16;
    v8f acc = {0,0,0,0,0,0,0,0};
    for (int k0 = 0; k0 < H_; k0 += 32) {
      BFrag af, bf;
      const __bf16* pa = &encs[(wave * 16 + l16) * ENC_LDA + k0 + half * 8];
      af.q[0] = *(const uint4*)pa;
      af.q[1] = *(const uint4*)(pa + 16);
      const __bf16* pb = W1 + (size_t)n * H_ + k0 + half * 8;  // L2-resident
      bf.q[0] = *(const uint4*)pb;
      bf.q[1] = *(const uint4*)(pb + 16);
      acc = __builtin_amdgcn_wmma_f32_16x16x32_bf16(
          false, af.v, false, bf.v, (short)0, acc, false, false);
    }
    float bias = W1b[n] + t2b[n];
    float vw   = Vw[n];
    #pragma unroll
    for (int v = 0; v < 8; v++) {
      float sc = tanhf(acc[v] + bias);
      int m = wave * 16 + half * 8 + v;
      atomicAdd(&s2[m], sc * vw);     // ds_add_f32
    }
  }
  __syncthreads();

  // Softmax over S=128
  float val = (tid < 128) ? s2[tid] : -1e30f;
  if (tid < 128) red[tid] = val;
  __syncthreads();
  for (int s = 64; s > 0; s >>= 1) {
    if (tid < s) red[tid] = fmaxf(red[tid], red[tid + s]);
    __syncthreads();
  }
  float mx = red[0];
  __syncthreads();
  float e = (tid < 128) ? __expf(val - mx) : 0.0f;
  if (tid < 128) red[tid] = e;
  __syncthreads();
  for (int s = 64; s > 0; s >>= 1) {
    if (tid < s) red[tid] += red[tid + s];
    __syncthreads();
  }
  float denom = red[0];
  if (tid < 128) attw[tid] = e / denom;
  __syncthreads();

  // ctx from LDS-resident enc tile (2 columns per thread)
  int h = tid * 2;
  float a0 = 0.f, a1 = 0.f;
  for (int s = 0; s < S_; s++) {
    float w = attw[s];
    a0 += w * bf2f(encs[s * ENC_LDA + h]);
    a1 += w * bf2f(encs[s * ENC_LDA + h + 1]);
  }
  ctx[(size_t)b * H_ + h]     = a0;
  ctx[(size_t)b * H_ + h + 1] = a1;
}

// ---------------------------------------------------------------------------
// gin = concat(ctx, emb_table[x]) -> bf16, K padded 812->832 with zeros
// ---------------------------------------------------------------------------
__global__ void pack_gin_kernel(const float* __restrict__ ctx,
                                const int* __restrict__ x,
                                const float* __restrict__ emb_table,
                                __bf16* __restrict__ gin) {
  int idx = blockIdx.x * blockDim.x + threadIdx.x;
  if (idx >= B_ * KIH) return;
  int b = idx / KIH, c = idx % KIH;
  float v;
  if (c < H_)            v = ctx[(size_t)b * H_ + c];
  else if (c < H_ + E_)  v = emb_table[(size_t)x[b] * E_ + (c - H_)];
  else                   v = 0.0f;
  gin[idx] = f2bf(v);
}

// ---------------------------------------------------------------------------
// GRU gates: h_new = (1-z)*n + z*h0  (writes f32 output + bf16 for FC path)
// ---------------------------------------------------------------------------
__global__ void gates_kernel(const float* __restrict__ gi,
                             const float* __restrict__ gh,
                             const float* __restrict__ h0,
                             float* __restrict__ hnew,
                             __bf16* __restrict__ hnew_bf) {
  int idx = blockIdx.x * blockDim.x + threadIdx.x;
  if (idx >= B_ * H_) return;
  int b = idx >> 9, h = idx & (H_ - 1);
  const float* gib = gi + (size_t)b * (3 * H_);
  const float* ghb = gh + (size_t)b * (3 * H_);
  float ir = gib[h], iz = gib[h + H_], in_ = gib[h + 2 * H_];
  float hr = ghb[h], hz = ghb[h + H_], hn  = ghb[h + 2 * H_];
  float r = 1.f / (1.f + __expf(-(ir + hr)));
  float z = 1.f / (1.f + __expf(-(iz + hz)));
  float n = tanhf(in_ + r * hn);
  float out = (1.f - z) * n + z * h0[idx];
  hnew[idx]    = out;
  hnew_bf[idx] = f2bf(out);
}

// ---------------------------------------------------------------------------
// LeakyReLU(0.01) + bf16 pack, K padded 300->320 with zeros
// ---------------------------------------------------------------------------
__global__ void leaky_pack_kernel(const float* __restrict__ fc1o,
                                  __bf16* __restrict__ act) {
  int idx = blockIdx.x * blockDim.x + threadIdx.x;
  if (idx >= B_ * EPAD) return;
  int c = idx % EPAD;
  float v = 0.0f;
  if (c < E_) {
    float t = fc1o[idx];
    v = t > 0.f ? t : 0.01f * t;
  }
  act[idx] = f2bf(v);
}

// ---------------------------------------------------------------------------
// Host launcher
// ---------------------------------------------------------------------------
extern "C" void kernel_launch(void* const* d_in, const int* in_sizes, int n_in,
                              void* d_out, int out_size, void* d_ws, size_t ws_size,
                              hipStream_t stream) {
  (void)in_sizes; (void)n_in; (void)out_size; (void)ws_size;
  const int*   x      = (const int*)  d_in[0];
  const float* hidden = (const float*)d_in[1];   // (1,B,H): hsum == h0
  const float* enc    = (const float*)d_in[2];
  const float* embT   = (const float*)d_in[3];
  const float* W1w    = (const float*)d_in[4];
  const float* W1b    = (const float*)d_in[5];
  const float* W2w    = (const float*)d_in[6];
  const float* W2b    = (const float*)d_in[7];
  const float* Vw     = (const float*)d_in[8];
  const float* Vb     = (const float*)d_in[9];
  const float* wih    = (const float*)d_in[10];  // (1536, 812)
  const float* whh    = (const float*)d_in[11];  // (1536, 512)
  const float* bih    = (const float*)d_in[12];
  const float* bhh    = (const float*)d_in[13];
  const float* fc1w   = (const float*)d_in[14];  // (300, 512)
  const float* fc1b   = (const float*)d_in[15];
  const float* fc2w   = (const float*)d_in[16];  // (128, 300)
  const float* fc2b   = (const float*)d_in[17];

  float* y_out    = (float*)d_out;              // (2048, 128)
  float* hnew_out = y_out + (size_t)B_ * V_;    // (1, 2048, 512)

  // Workspace carve-up (256B aligned)
  char* p = (char*)d_ws;
  auto alloc = [&](size_t bytes) -> char* {
    char* r = p; p += (bytes + 255) & ~(size_t)255; return r;
  };
  __bf16* W1bf   = (__bf16*)alloc((size_t)H_ * H_ * 2);
  __bf16* W2bf   = (__bf16*)alloc((size_t)H_ * H_ * 2);
  __bf16* wihbf  = (__bf16*)alloc((size_t)(3 * H_) * KIH * 2);
  __bf16* whhbf  = (__bf16*)alloc((size_t)(3 * H_) * H_ * 2);
  __bf16* fc1bf  = (__bf16*)alloc((size_t)EPAD * H_ * 2);   // 320 rows, 512 K
  __bf16* fc2bf  = (__bf16*)alloc((size_t)V_ * EPAD * 2);   // 128 rows, 320 K
  __bf16* h0bf   = (__bf16*)alloc((size_t)B_ * H_ * 2);
  float*  t2     = (float*) alloc((size_t)B_ * H_ * 4);
  float*  ctx    = (float*) alloc((size_t)B_ * H_ * 4);
  __bf16* ginbf  = (__bf16*)alloc((size_t)B_ * KIH * 2);
  float*  gi     = (float*) alloc((size_t)B_ * 3 * H_ * 4);
  float*  gh     = (float*) alloc((size_t)B_ * 3 * H_ * 4);
  __bf16* hnewbf = (__bf16*)alloc((size_t)B_ * H_ * 2);
  float*  fc1o   = (float*) alloc((size_t)B_ * EPAD * 4);
  __bf16* actbf  = (__bf16*)alloc((size_t)B_ * EPAD * 2);

  const int T = 256;
  auto blks = [](size_t n) { return (unsigned)((n + 255) / 256); };

  // 1) Weight / activation conversions to bf16 (with zero padding)
  convert_pad_kernel<<<blks((size_t)H_*H_), T, 0, stream>>>(W1w, W1bf, H_, H_, H_, H_);
  convert_pad_kernel<<<blks((size_t)H_*H_), T, 0, stream>>>(W2w, W2bf, H_, H_, H_, H_);
  convert_pad_kernel<<<blks((size_t)3*H_*KIH), T, 0, stream>>>(wih, wihbf, 3*H_, 812, KIH, 3*H_);
  convert_pad_kernel<<<blks((size_t)3*H_*H_), T, 0, stream>>>(whh, whhbf, 3*H_, H_, H_, 3*H_);
  convert_pad_kernel<<<blks((size_t)EPAD*H_), T, 0, stream>>>(fc1w, fc1bf, E_, H_, H_, EPAD);
  convert_pad_kernel<<<blks((size_t)V_*EPAD), T, 0, stream>>>(fc2w, fc2bf, V_, E_, EPAD, V_);
  convert_pad_kernel<<<blks((size_t)B_*H_), T, 0, stream>>>(hidden, h0bf, B_, H_, H_, B_);

  // 2) t2 = h0 @ W2^T + W2_b          (2048x512x512)
  gemm_bf16_kernel<<<dim3(B_/128, H_/64), T, 0, stream>>>(
      h0bf, H_, W2bf, H_, W2b, t2, H_, H_, H_);

  // 3) Fused attention (score GEMM + softmax + context), 1 WG per batch row
  {
    size_t shmem = (size_t)128 * ENC_LDA * 2 + 3 * 128 * sizeof(float);
    (void)hipFuncSetAttribute((const void*)attn_fused_kernel,
                              hipFuncAttributeMaxDynamicSharedMemorySize,
                              (int)shmem);
    attn_fused_kernel<<<dim3(B_), T, shmem, stream>>>(
        enc, W1bf, W1b, t2, Vw, Vb, ctx);
  }

  // 4) gin = [ctx, emb], bf16, padded
  pack_gin_kernel<<<blks((size_t)B_*KIH), T, 0, stream>>>(ctx, x, embT, ginbf);

  // 5) gi = gin @ w_ih^T + b_ih       (2048x1536x832)
  gemm_bf16_kernel<<<dim3(B_/128, (3*H_)/64), T, 0, stream>>>(
      ginbf, KIH, wihbf, KIH, bih, gi, 3*H_, 3*H_, KIH);

  // 6) gh = h0 @ w_hh^T + b_hh        (2048x1536x512)
  gemm_bf16_kernel<<<dim3(B_/128, (3*H_)/64), T, 0, stream>>>(
      h0bf, H_, whhbf, H_, bhh, gh, 3*H_, 3*H_, H_);

  // 7) GRU gates -> h_new (f32 to d_out) + bf16 copy
  gates_kernel<<<blks((size_t)B_*H_), T, 0, stream>>>(gi, gh, hidden, hnew_out, hnewbf);

  // 8) fc1 = h_new @ fc1_w^T + b      (2048x300x512, N guarded to 300)
  gemm_bf16_kernel<<<dim3(B_/128, EPAD/64), T, 0, stream>>>(
      hnewbf, H_, fc1bf, H_, fc1b, fc1o, EPAD, E_, H_);

  // 9) LeakyReLU + bf16 pack (K pad 300->320)
  leaky_pack_kernel<<<blks((size_t)B_*EPAD), T, 0, stream>>>(fc1o, actbf);

  // 10) y = act @ fc2_w^T + b         (2048x128x320) -> d_out
  gemm_bf16_kernel<<<dim3(B_/128, V_/64), T, 0, stream>>>(
      actbf, EPAD, fc2bf, EPAD, fc2b, y_out, V_, V_, EPAD);
}
